// HopfieldPoolingRouter_55757265436956
// MI455X (gfx1250) — matrix operand order repaired and checked by reading, not compiled, not run on hardware
//
#include <hip/hip_runtime.h>
#include <cstdint>
#include <cstddef>

// ---------------- problem constants ----------------
#define B_SZ   128
#define N_SZ   32768
#define D_SZ   768
#define H_SZ   16
#define DH     48
#define DHP    64                 // per-head padded to 64 for 16x16x32 WMMA (2 k-chunks)
#define KSTR   (H_SZ * DHP)       // 1024: padded row stride of k/v/xi
#define BETA   0.25f
#define NSPLIT 32                 // attention key-splits per head
#define KEYS_PER_SPLIT (N_SZ / NSPLIT)   // 1024

typedef __attribute__((ext_vector_type(16))) __bf16 v16bf;
typedef __attribute__((ext_vector_type(8)))  float  v8f;
typedef __attribute__((ext_vector_type(4)))  unsigned int uint4v;

union bf8u  { uint4v u;    __bf16 b[8]; };   // 8 bf16 = one 16B vector
union frag16 { v16bf v;    uint4v u[2]; };   // 16 bf16 = two 16B vectors

static __device__ inline v8f v8f_zero() {
  v8f z;
#pragma unroll
  for (int i = 0; i < 8; ++i) z[i] = 0.0f;
  return z;
}

static __device__ inline v8f wmma_bf16(v16bf a, v16bf b, v8f c) {
  // D = A(16x32 bf16) x B(32x16 bf16) + C(16x16 f32)
  return __builtin_amdgcn_wmma_f32_16x16x32_bf16(false, a, false, b, (short)0, c,
                                                 false, false);
}

// A fragment (16x32 bf16) from a row-major tile (LDS or global), ld in elements.
// lane l: row m=l&15, k-half kh=l>>4; one contiguous 16-element segment:
//   k in [16kh, 16kh+16)  ->  2 x b128 loads
static __device__ inline v16bf load_a_frag(const __bf16* s, int ld) {
  const int l = threadIdx.x & 31;
  const int m = l & 15, kh = l >> 4;
  const __bf16* p = s + m * ld + kh * 16;
  frag16 f;
  f.u[0] = *(const uint4v*)p;
  f.u[1] = *(const uint4v*)(p + 8);
  return f.v;
}

// 16 contiguous bf16 -> B fragment lane data (2 x b128)
static __device__ inline v16bf load_b16(const __bf16* p) {
  frag16 f;
  f.u[0] = *(const uint4v*)p;
  f.u[1] = *(const uint4v*)(p + 8);
  return f.v;
}

// ---------------- per-row stats: LN mean/rstd (biased var) + inverse L2 norm ---
__global__ __launch_bounds__(256) void row_stats_kernel(
    const float* __restrict__ src, int nrows,
    float* __restrict__ mean, float* __restrict__ rstd, float* __restrict__ invn) {
  const int wave = threadIdx.x >> 5, lane = threadIdx.x & 31;
  const int row = blockIdx.x * 8 + wave;
  if (row >= nrows) return;
  const float* r = src + (size_t)row * D_SZ;
  float s = 0.f, sq = 0.f;
#pragma unroll
  for (int i = 0; i < D_SZ / 32; ++i) {
    const float x = r[lane + i * 32];
    s += x; sq += x * x;
  }
#pragma unroll
  for (int m = 1; m <= 16; m <<= 1) {
    s  += __shfl_xor(s,  m, 32);
    sq += __shfl_xor(sq, m, 32);
  }
  if (lane == 0) {
    const float mu  = s * (1.0f / D_SZ);
    const float var = sq * (1.0f / D_SZ) - mu * mu;
    mean[row] = mu;
    rstd[row] = rsqrtf(var + 1e-5f);
    invn[row] = 1.0f / fmaxf(sqrtf(sq), 1e-12f);
  }
}

// ---------------- projection GEMM: out = LN(src) @ W^T + bias, bf16 head-padded --
#define LDA 40
__global__ __launch_bounds__(256) void proj_kernel(
    const float* __restrict__ src, const float* __restrict__ mean,
    const float* __restrict__ rstd, const float* __restrict__ gamma,
    const float* __restrict__ beta, const float* __restrict__ W,
    const float* __restrict__ bias, __bf16* __restrict__ out) {
  __shared__ __align__(16) __bf16 As[128 * LDA];      // row-major A tile 128x32
  __shared__ __align__(16) __bf16 Bsw[4 * 512];       // frag-swizzled B: 4 subtiles
  const int t = threadIdx.x, wave = t >> 5, lane = t & 31;
  const int row0 = blockIdx.x * 128;
  const int j0 = blockIdx.y * 64;
  v8f acc[4];
#pragma unroll
  for (int i = 0; i < 4; ++i) acc[i] = v8f_zero();

  // A staging: 2 threads/row, 16 consecutive cols each
  const int arow = t >> 1, acol0 = (t & 1) * 16;
  const float mu = mean[row0 + arow], rs = rstd[row0 + arow];
  // B staging (swizzled): pair=(subtile,lane), each thread does 8 consecutive d
  const int bpair = t >> 1, bjh = (t & 1) * 8;
  const int bnt = bpair >> 5, bl = bpair & 31;
  const int bout = j0 + bnt * 16 + (bl & 15);          // output column
  const int bd0 = ((bl >> 4) << 4) + bjh;              // local d base

  float areg[16], garr[16], brr[16], breg[8];
  auto load_chunk = [&](int kc) {
    const float* sp = src + (size_t)(row0 + arow) * D_SZ + kc + acol0;
#pragma unroll
    for (int c = 0; c < 16; ++c) {
      areg[c] = sp[c];
      garr[c] = gamma[kc + acol0 + c];
      brr[c]  = beta[kc + acol0 + c];
    }
    const float* wp = W + (size_t)bout * D_SZ + kc + bd0;
#pragma unroll
    for (int c = 0; c < 8; ++c) breg[c] = wp[c];
  };
  load_chunk(0);

  for (int kc = 0; kc < D_SZ; kc += 32) {
    __syncthreads();
    {  // A tile store (LayerNorm fused), 2 x b128
      bf8u p0, p1;
#pragma unroll
      for (int c = 0; c < 8; ++c) {
        p0.b[c] = (__bf16)((areg[c] - mu) * rs * garr[c] + brr[c]);
        p1.b[c] = (__bf16)((areg[c + 8] - mu) * rs * garr[c + 8] + brr[c + 8]);
      }
      *(uint4v*)&As[arow * LDA + acol0]     = p0.u;
      *(uint4v*)&As[arow * LDA + acol0 + 8] = p1.u;
    }
    {  // B tile store into frag-swizzled layout, 1 x b128
      bf8u q;
#pragma unroll
      for (int c = 0; c < 8; ++c) q.b[c] = (__bf16)breg[c];
      *(uint4v*)&Bsw[bnt * 512 + bl * 16 + bjh] = q.u;
    }
    __syncthreads();
    if (kc + 32 < D_SZ) load_chunk(kc + 32);           // prefetch next chunk
    const v16bf a = load_a_frag(&As[wave * 16 * LDA], LDA);
#pragma unroll
    for (int nt = 0; nt < 4; ++nt) {
      const v16bf b = load_b16(&Bsw[nt * 512 + lane * 16]);
      acc[nt] = wmma_bf16(a, b, acc[nt]);
    }
  }
  // store: head-padded layout col j -> (j/48)*64 + j%48
  const int mb = (lane >> 4) * 8, n = lane & 15;
#pragma unroll
  for (int nt = 0; nt < 4; ++nt) {
    const int jg = j0 + nt * 16 + n;
    const int pj = (jg / DH) * DHP + (jg % DH);
    const float bvv = bias[jg];
#pragma unroll
    for (int r = 0; r < 8; ++r) {
      const int rg = row0 + wave * 16 + mb + r;
      out[(size_t)rg * KSTR + pj] = (__bf16)(acc[nt][r] + bvv);
    }
  }
}

// ---------------- small-M GEMM (M=128): out = act(A @ W^T + bias [+ res]) f32 ----
__global__ __launch_bounds__(256) void mlp_kernel(
    const float* __restrict__ A, const float* __restrict__ W,
    const float* __restrict__ bias, const float* __restrict__ residual,
    int do_relu, float* __restrict__ out) {
  __shared__ __align__(16) __bf16 As[128 * LDA];
  __shared__ __align__(16) __bf16 Bsw[4 * 512];
  const int t = threadIdx.x, wave = t >> 5, lane = t & 31;
  const int j0 = blockIdx.x * 64;
  v8f acc[4];
#pragma unroll
  for (int i = 0; i < 4; ++i) acc[i] = v8f_zero();
  const int arow = t >> 1, acol0 = (t & 1) * 16;
  const int bpair = t >> 1, bjh = (t & 1) * 8;
  const int bnt = bpair >> 5, bl = bpair & 31;
  const int bout = j0 + bnt * 16 + (bl & 15);
  const int bd0 = ((bl >> 4) << 4) + bjh;

  float areg[16], breg[8];
  auto load_chunk = [&](int kc) {
    const float* sp = A + (size_t)arow * D_SZ + kc + acol0;
#pragma unroll
    for (int c = 0; c < 16; ++c) areg[c] = sp[c];
    const float* wp = W + (size_t)bout * D_SZ + kc + bd0;
#pragma unroll
    for (int c = 0; c < 8; ++c) breg[c] = wp[c];
  };
  load_chunk(0);

  for (int kc = 0; kc < D_SZ; kc += 32) {
    __syncthreads();
    {
      bf8u p0, p1;
#pragma unroll
      for (int c = 0; c < 8; ++c) {
        p0.b[c] = (__bf16)areg[c];
        p1.b[c] = (__bf16)areg[c + 8];
      }
      *(uint4v*)&As[arow * LDA + acol0]     = p0.u;
      *(uint4v*)&As[arow * LDA + acol0 + 8] = p1.u;
    }
    {
      bf8u q;
#pragma unroll
      for (int c = 0; c < 8; ++c) q.b[c] = (__bf16)breg[c];
      *(uint4v*)&Bsw[bnt * 512 + bl * 16 + bjh] = q.u;
    }
    __syncthreads();
    if (kc + 32 < D_SZ) load_chunk(kc + 32);
    const v16bf a = load_a_frag(&As[wave * 16 * LDA], LDA);
#pragma unroll
    for (int nt = 0; nt < 4; ++nt) {
      const v16bf b = load_b16(&Bsw[nt * 512 + lane * 16]);
      acc[nt] = wmma_bf16(a, b, acc[nt]);
    }
  }
  const int mb = (lane >> 4) * 8, n = lane & 15;
#pragma unroll
  for (int nt = 0; nt < 4; ++nt) {
    const int jg = j0 + nt * 16 + n;
    const float bvv = bias[jg];
#pragma unroll
    for (int r = 0; r < 8; ++r) {
      const int rg = wave * 16 + mb + r;
      float v = acc[nt][r] + bvv;
      if (residual) v += residual[(size_t)rg * D_SZ + jg];
      if (do_relu) v = fmaxf(v, 0.0f);
      out[(size_t)rg * D_SZ + jg] = v;
    }
  }
}

// ---------------- flash attention partial: one (head, key-split) per block ------
// scores = beta * xi @ k^T ; online softmax ; partial numerator = P @ agg
__global__ __launch_bounds__(256) void attn_kernel(
    const __bf16* __restrict__ xi, const __bf16* __restrict__ kp,
    const __bf16* __restrict__ agg, float* __restrict__ pnum,
    float* __restrict__ pmax, float* __restrict__ psum) {
  __shared__ __align__(16) __bf16 Kt[64 * 72];     // raw k tile (keys x dh) for scores
  __shared__ __align__(16) __bf16 Aggsw[8 * 512];  // frag-swizzled agg tile
  __shared__ __align__(16) __bf16 Pst[8][16 * 72]; // per-wave probability staging
  const int t = threadIdx.x, wave = t >> 5, lane = t & 31;
  const int split = blockIdx.x, h = blockIdx.y;
  const int brow0 = wave * 16;
  const int mb = (lane >> 4) * 8, nn = lane & 15, kb = (lane >> 4) << 4;

  // preload xi A-fragments for k-chunks 0,1 (fixed over the key loop)
  v16bf axi[2];
#pragma unroll
  for (int c = 0; c < 2; ++c)
    axi[c] = load_a_frag(xi + (size_t)brow0 * KSTR + h * DHP + c * 32, KSTR);

  // cooperative staging indices: 64 keys x 64 dh, 16 elements per thread
  const int skey = t >> 2, sdh = (t & 3) * 16;
  const int sct = skey >> 5, slh = (skey >> 4) & 1, sjj = skey & 15, sdt = t & 3;

  float mrun[8], srun[8];
#pragma unroll
  for (int r = 0; r < 8; ++r) { mrun[r] = -3.0e38f; srun[r] = 0.0f; }
  v8f acc[4];
#pragma unroll
  for (int i = 0; i < 4; ++i) acc[i] = v8f_zero();

  for (int kt = 0; kt < KEYS_PER_SPLIT / 64; ++kt) {
    const int key0 = split * KEYS_PER_SPLIT + kt * 64;
    __builtin_prefetch(kp + (size_t)(key0 + 64) * KSTR + h * DHP, 0, 0);

    __syncthreads();
    {  // k tile: raw row-major (contiguous 32B per thread, in and out)
      const __bf16* ksrc = kp + (size_t)(key0 + skey) * KSTR + h * DHP + sdh;
      uint4v k0 = *(const uint4v*)ksrc;
      uint4v k1 = *(const uint4v*)(ksrc + 8);
      *(uint4v*)&Kt[skey * 72 + sdh]     = k0;
      *(uint4v*)&Kt[skey * 72 + sdh + 8] = k1;
      // agg tile: scatter into frag-swizzled layout
      const __bf16* asrc = agg + (size_t)(key0 + skey) * KSTR + h * DHP + sdh;
      bf8u a0, a1;
      a0.u = *(const uint4v*)asrc;
      a1.u = *(const uint4v*)(asrc + 8);
      __bf16* dst = &Aggsw[(sct * 4 + sdt) * 512 + sjj];
#pragma unroll
      for (int n = 0; n < 8; ++n) {
        dst[(slh * 16 + n) * 16]     = a0.b[n];
        dst[(slh * 16 + 8 + n) * 16] = a1.b[n];
      }
    }
    __syncthreads();

    // scores: 16 batch rows x 64 keys (transposed B-frags = contiguous Kt rows)
    v8f sc[4];
#pragma unroll
    for (int i = 0; i < 4; ++i) sc[i] = v8f_zero();
#pragma unroll
    for (int st = 0; st < 4; ++st) {
#pragma unroll
      for (int c = 0; c < 2; ++c) {
        const v16bf b = load_b16(&Kt[(st * 16 + nn) * 72 + c * 32 + kb]);
        sc[st] = wmma_bf16(axi[c], b, sc[st]);
      }
    }

    // online softmax over this 64-key tile (row r lives in element r: lane-local)
#pragma unroll
    for (int r = 0; r < 8; ++r) {
      float v = fmaxf(fmaxf(sc[0][r], sc[1][r]), fmaxf(sc[2][r], sc[3][r])) * BETA;
#pragma unroll
      for (int msk = 1; msk <= 8; msk <<= 1) v = fmaxf(v, __shfl_xor(v, msk, 32));
      const float mnew = fmaxf(mrun[r], v);
      const float scale = __expf(mrun[r] - mnew);
      float lsum = 0.f;
#pragma unroll
      for (int st = 0; st < 4; ++st) {
        const float pv = __expf(sc[st][r] * BETA - mnew);
        lsum += pv;
        Pst[wave][(mb + r) * 72 + st * 16 + nn] = (__bf16)pv;
      }
#pragma unroll
      for (int msk = 1; msk <= 8; msk <<= 1) lsum += __shfl_xor(lsum, msk, 32);
      srun[r] = srun[r] * scale + lsum;
      mrun[r] = mnew;
#pragma unroll
      for (int dt = 0; dt < 4; ++dt) acc[dt][r] *= scale;
    }

    // numerator: P(16x64) @ agg-rows(64 x DHP), swizzled B-frags (2 x b128 each)
#pragma unroll
    for (int c = 0; c < 2; ++c) {
      const v16bf pa = load_a_frag(&Pst[wave][c * 32], 72);
#pragma unroll
      for (int dt = 0; dt < 4; ++dt) {
        const v16bf bg = load_b16(&Aggsw[(c * 4 + dt) * 512 + lane * 16]);
        acc[dt] = wmma_bf16(pa, bg, acc[dt]);
      }
    }
  }

  // write partials for the combine pass
  const size_t base = ((size_t)h * NSPLIT + split) * B_SZ;
#pragma unroll
  for (int dt = 0; dt < 4; ++dt)
#pragma unroll
    for (int r = 0; r < 8; ++r)
      pnum[(base + brow0 + mb + r) * DHP + dt * 16 + nn] = acc[dt][r];
  if (nn == 0) {
#pragma unroll
    for (int r = 0; r < 8; ++r) {
      pmax[base + brow0 + mb + r] = mrun[r];
      psum[base + brow0 + mb + r] = srun[r];
    }
  }
}

// ---------------- combine split partials -> xi_next (bf16) or attn out (f32) ----
__global__ __launch_bounds__(64) void combine_kernel(
    const float* __restrict__ pnum, const float* __restrict__ pmax,
    const float* __restrict__ psum, __bf16* __restrict__ xi_out,
    float* __restrict__ att_out, int final_step) {
  const int b = blockIdx.x, h = blockIdx.y, d = threadIdx.x;
  float M = -3.0e38f;
  for (int s = 0; s < NSPLIT; ++s)
    M = fmaxf(M, pmax[((size_t)h * NSPLIT + s) * B_SZ + b]);
  float num = 0.f, den = 0.f;
  for (int s = 0; s < NSPLIT; ++s) {
    const size_t idx = ((size_t)h * NSPLIT + s) * B_SZ + b;
    const float w = __expf(pmax[idx] - M);
    num += w * pnum[idx * DHP + d];
    den += w * psum[idx];
  }
  const float o = num / den;
  if (final_step) {
    if (d < DH) att_out[(size_t)b * D_SZ + h * DH + d] = o;
  } else {
    xi_out[(size_t)b * KSTR + h * DHP + d] = (__bf16)o;
  }
}

// ---------------- logits = 20 * l2norm(rq) @ l2norm(mem)^T ----------------------
__global__ __launch_bounds__(256) void logits_kernel(
    const float* __restrict__ rq, const float* __restrict__ rqInv,
    const float* __restrict__ memb, const float* __restrict__ memInv,
    float* __restrict__ logits) {
  __shared__ __align__(16) __bf16 As[128 * LDA];
  __shared__ __align__(16) __bf16 Bsw[8 * 512];
  const int t = threadIdx.x, wave = t >> 5, lane = t & 31;
  const int n0 = blockIdx.x * 128;
  v8f acc[8];
#pragma unroll
  for (int i = 0; i < 8; ++i) acc[i] = v8f_zero();
  const int arow = t >> 1, acol0 = (t & 1) * 16;
  const float iv = rqInv[arow];
  // B staging: one (subtile, lane) pair per thread, 16 consecutive d
  const int bnt = t >> 5, bl = t & 31;
  const int brow = n0 + bnt * 16 + (bl & 15);
  const int bd0 = (bl >> 4) << 4;
  const float biv = memInv[brow];

  float areg[16], breg[16];
  auto load_chunk = [&](int kc) {
    const float* sp = rq + (size_t)arow * D_SZ + kc + acol0;
#pragma unroll
    for (int c = 0; c < 16; ++c) areg[c] = sp[c];
    const float* mp = memb + (size_t)brow * D_SZ + kc + bd0;
#pragma unroll
    for (int c = 0; c < 16; ++c) breg[c] = mp[c];
  };
  load_chunk(0);

  for (int kc = 0; kc < D_SZ; kc += 32) {
    __syncthreads();
    {
      bf8u p0, p1;
#pragma unroll
      for (int c = 0; c < 8; ++c) {
        p0.b[c] = (__bf16)(areg[c] * iv);
        p1.b[c] = (__bf16)(areg[c + 8] * iv);
      }
      *(uint4v*)&As[arow * LDA + acol0]     = p0.u;
      *(uint4v*)&As[arow * LDA + acol0 + 8] = p1.u;
    }
    {
      bf8u q0, q1;
#pragma unroll
      for (int c = 0; c < 8; ++c) {
        q0.b[c] = (__bf16)(breg[c] * biv);
        q1.b[c] = (__bf16)(breg[c + 8] * biv);
      }
      *(uint4v*)&Bsw[bnt * 512 + bl * 16]     = q0.u;
      *(uint4v*)&Bsw[bnt * 512 + bl * 16 + 8] = q1.u;
    }
    __syncthreads();
    if (kc + 32 < D_SZ) load_chunk(kc + 32);
    const v16bf a = load_a_frag(&As[wave * 16 * LDA], LDA);
#pragma unroll
    for (int nt = 0; nt < 8; ++nt) {
      const v16bf b = load_b16(&Bsw[nt * 512 + lane * 16]);
      acc[nt] = wmma_bf16(a, b, acc[nt]);
    }
  }
  const int mb = (lane >> 4) * 8, nn = lane & 15;
#pragma unroll
  for (int nt = 0; nt < 8; ++nt)
#pragma unroll
    for (int r = 0; r < 8; ++r)
      logits[(size_t)(wave * 16 + mb + r) * N_SZ + n0 + nt * 16 + nn] =
          20.0f * acc[nt][r];
}

// ---------------- top-4 per row (iterated argmax, lowest index wins ties) -------
__global__ __launch_bounds__(256) void topk_kernel(const float* __restrict__ logits,
                                                   float* __restrict__ out_idx) {
  __shared__ float sval[256];
  __shared__ int   sidx[256];
  __shared__ int   chosen[4];
  const int b = blockIdx.x, t = threadIdx.x;
  const float* row = logits + (size_t)b * N_SZ;
  for (int it = 0; it < 4; ++it) {
    float bv = -3.0e38f; int bi = 0x7fffffff;
    for (int i = t; i < N_SZ; i += 256) {
      bool skip = false;
      for (int k = 0; k < it; ++k) if (chosen[k] == i) skip = true;
      const float v = row[i];
      if (!skip && (v > bv || (v == bv && i < bi))) { bv = v; bi = i; }
    }
    sval[t] = bv; sidx[t] = bi;
    __syncthreads();
    for (int s = 128; s > 0; s >>= 1) {
      if (t < s) {
        if (sval[t + s] > sval[t] ||
            (sval[t + s] == sval[t] && sidx[t + s] < sidx[t])) {
          sval[t] = sval[t + s]; sidx[t] = sidx[t + s];
        }
      }
      __syncthreads();
    }
    if (t == 0) { chosen[it] = sidx[0]; out_idx[b * 4 + it] = (float)sidx[0]; }
    __syncthreads();
  }
}

// ------------------------------- launcher ---------------------------------------
extern "C" void kernel_launch(void* const* d_in, const int* in_sizes, int n_in,
                              void* d_out, int out_size, void* d_ws, size_t ws_size,
                              hipStream_t stream) {
  (void)in_sizes; (void)n_in; (void)out_size; (void)ws_size;
  const float* query  = (const float*)d_in[0];
  const float* memb   = (const float*)d_in[1];
  const float* ln_k_g = (const float*)d_in[2];
  const float* ln_k_b = (const float*)d_in[3];
  const float* ln_q_g = (const float*)d_in[4];
  const float* ln_q_b = (const float*)d_in[5];
  const float* ln_v_g = (const float*)d_in[6];
  const float* ln_v_b = (const float*)d_in[7];
  const float* Wq = (const float*)d_in[8];  const float* bq = (const float*)d_in[9];
  const float* Wk = (const float*)d_in[10]; const float* bk = (const float*)d_in[11];
  const float* Wv = (const float*)d_in[12]; const float* bvv = (const float*)d_in[13];
  const float* Wo = (const float*)d_in[14]; const float* bo = (const float*)d_in[15];
  const float* W1 = (const float*)d_in[16]; const float* b1 = (const float*)d_in[17];
  const float* W2 = (const float*)d_in[18]; const float* b2 = (const float*)d_in[19];

  float* out      = (float*)d_out;
  float* out_idx  = out;                    // B*TOPK indices (as float)
  float* logits   = out + B_SZ * 4;         // B*N logits

  // ---- carve workspace ----
  uint8_t* p = (uint8_t*)d_ws;
  auto carve = [&](size_t bytes) -> void* {
    void* r = (void*)p;
    p += (bytes + 255) & ~(size_t)255;
    return r;
  };
  __bf16* kpad = (__bf16*)carve((size_t)N_SZ * KSTR * sizeof(__bf16));
  __bf16* vpad = (__bf16*)carve((size_t)N_SZ * KSTR * sizeof(__bf16));
  __bf16* xiA  = (__bf16*)carve((size_t)B_SZ * KSTR * sizeof(__bf16));
  __bf16* xiB  = (__bf16*)carve((size_t)B_SZ * KSTR * sizeof(__bf16));
  float* memMean = (float*)carve((size_t)N_SZ * 4);
  float* memRstd = (float*)carve((size_t)N_SZ * 4);
  float* memInv  = (float*)carve((size_t)N_SZ * 4);
  float* qMean   = (float*)carve((size_t)B_SZ * 4);
  float* qRstd   = (float*)carve((size_t)B_SZ * 4);
  float* qInv    = (float*)carve((size_t)B_SZ * 4);
  float* rqMean  = (float*)carve((size_t)B_SZ * 4);
  float* rqRstd  = (float*)carve((size_t)B_SZ * 4);
  float* rqInv   = (float*)carve((size_t)B_SZ * 4);
  float* pnum = (float*)carve((size_t)H_SZ * NSPLIT * B_SZ * DHP * 4);
  float* pmax = (float*)carve((size_t)H_SZ * NSPLIT * B_SZ * 4);
  float* psum = (float*)carve((size_t)H_SZ * NSPLIT * B_SZ * 4);
  float* attnout  = (float*)carve((size_t)B_SZ * D_SZ * 4);
  float* combined = (float*)carve((size_t)B_SZ * D_SZ * 4);
  float* hid      = (float*)carve((size_t)B_SZ * D_SZ * 4);
  float* rqb      = (float*)carve((size_t)B_SZ * D_SZ * 4);

  // zero head-pad lanes so padded columns contribute exact 0 to WMMA
  hipMemsetAsync(kpad, 0, (size_t)N_SZ * KSTR * sizeof(__bf16), stream);
  hipMemsetAsync(vpad, 0, (size_t)N_SZ * KSTR * sizeof(__bf16), stream);
  hipMemsetAsync(xiA,  0, (size_t)B_SZ * KSTR * sizeof(__bf16), stream);

  // row statistics (LN mean/rstd shared by ln_k & ln_v; inv L2 norm for logits)
  row_stats_kernel<<<N_SZ / 8, 256, 0, stream>>>(memb, N_SZ, memMean, memRstd, memInv);
  row_stats_kernel<<<B_SZ / 8, 256, 0, stream>>>(query, B_SZ, qMean, qRstd, qInv);

  // projections (LN fused into A-tile load), bf16 head-padded outputs
  dim3 pgrid(N_SZ / 128, D_SZ / 64);
  proj_kernel<<<pgrid, 256, 0, stream>>>(memb, memMean, memRstd, ln_k_g, ln_k_b,
                                         Wk, bk, kpad);
  proj_kernel<<<pgrid, 256, 0, stream>>>(memb, memMean, memRstd, ln_v_g, ln_v_b,
                                         Wv, bvv, vpad);
  proj_kernel<<<dim3(1, D_SZ / 64), 256, 0, stream>>>(query, qMean, qRstd, ln_q_g,
                                                      ln_q_b, Wq, bq, xiA);

  // 5 Hopfield steps (aggregate k) + final readout (aggregate v)
  for (int step = 0; step <= 5; ++step) {
    const __bf16* xin  = (step & 1) ? xiB : xiA;
    __bf16* xout       = (step & 1) ? xiA : xiB;
    const __bf16* agg  = (step < 5) ? kpad : vpad;
    attn_kernel<<<dim3(NSPLIT, H_SZ), 256, 0, stream>>>(xin, kpad, agg, pnum, pmax,
                                                        psum);
    combine_kernel<<<dim3(B_SZ, H_SZ), 64, 0, stream>>>(pnum, pmax, psum, xout,
                                                        attnout, step == 5);
  }

  // retrieval head
  mlp_kernel<<<D_SZ / 64, 256, 0, stream>>>(attnout, Wo, bo, query, 0, combined);
  mlp_kernel<<<D_SZ / 64, 256, 0, stream>>>(combined, W1, b1, nullptr, 1, hid);
  mlp_kernel<<<D_SZ / 64, 256, 0, stream>>>(hid, W2, b2, nullptr, 0, rqb);

  row_stats_kernel<<<B_SZ / 8, 256, 0, stream>>>(rqb, B_SZ, rqMean, rqRstd, rqInv);
  logits_kernel<<<N_SZ / 128, 256, 0, stream>>>(rqb, rqInv, memb, memInv, logits);
  topk_kernel<<<B_SZ, 256, 0, stream>>>(logits, out_idx);
}
